// MultiHeadAttention_66726611911507
// MI455X (gfx1250) — matrix operand order repaired and checked by reading, not compiled, not run on hardware
//
#include <hip/hip_runtime.h>

// ---------------- problem constants ----------------
constexpr int Bn    = 4;
constexpr int T     = 2048;
constexpr int TQ    = T - 1;      // 2047
constexpr int DM    = 512;
constexpr int H     = 8;
constexpr int DEPTH = 64;
constexpr float SLOPE  = 0.2f;
constexpr float NEGINF = -1e9f;

typedef __attribute__((ext_vector_type(16))) _Float16 v16h;
typedef __attribute__((ext_vector_type(8)))  float    v8f;
typedef unsigned int u32x4 __attribute__((ext_vector_type(4)));
typedef int          i32x4 __attribute__((ext_vector_type(4)));
typedef int          i32x8 __attribute__((ext_vector_type(8)));

// ---------------- WMMA fragment helpers (ISA 7.12.2 layouts, wave32) ----------------
// A 16x32 f16: lane l (l16 = row M), hi = l>>4 selects K interleave:
//   frag[0..7]  = K = hi*8 + 0..7 ;  frag[8..15] = K = 16 + hi*8 + 0..7
__device__ __forceinline__ v16h frag_a_row_f16(const _Float16* rowp, int hi) {
  v16h a;
#pragma unroll
  for (int j = 0; j < 8; ++j) {
    a[j]     = rowp[hi * 8 + j];
    a[j + 8] = rowp[16 + hi * 8 + j];
  }
  return a;
}

// From f32 source (converted on load), zero-filled past kRem.
__device__ __forceinline__ v16h frag_a_row_f32(const float* rowp, int hi, int kRem) {
  v16h a;
#pragma unroll
  for (int j = 0; j < 8; ++j) {
    int ka = hi * 8 + j;
    int kb = 16 + hi * 8 + j;
    a[j]     = (ka < kRem) ? (_Float16)rowp[ka] : (_Float16)0.0f;
    a[j + 8] = (kb < kRem) ? (_Float16)rowp[kb] : (_Float16)0.0f;
  }
  return a;
}

// B 32x16 f16 from column-major storage: colp points at column N's K-run.
// lane l: column N = l16, frag[0..7] = K = k0 + hi*8 + 0..7 ; frag[8..15] = K = k0 + 16 + hi*8 + 0..7
__device__ __forceinline__ v16h frag_b_col_f16(const _Float16* colp, int k0, int hi) {
  v16h b;
#pragma unroll
  for (int j = 0; j < 8; ++j) { b[j] = colp[k0 + hi * 8 + j]; b[j + 8] = colp[k0 + 16 + hi * 8 + j]; }
  return b;
}

__device__ __forceinline__ v8f wmma_f32(v16h a, v16h b, v8f c) {
  return __builtin_amdgcn_wmma_f32_16x16x32_f16(false, a, false, b, (short)0, c,
                                                false, false);
}

// ---------------- weight convert+transpose: f32 row-major [K x N] -> f16 col-major [N][K] -------
__global__ void w_to_f16t(const float* __restrict__ W, _Float16* __restrict__ Wt,
                          int K, int N) {
  int idx = blockIdx.x * blockDim.x + threadIdx.x;
  if (idx >= K * N) return;
  int n = idx / K;
  int k = idx % K;
  Wt[idx] = (_Float16)W[(long)k * N + n];
}

// ---------------- mask (f32 0/1, 67MB) -> bitmask (2MB, L2-resident for all 8 heads) ----------
__global__ void mask_to_bits(const float* __restrict__ mask, unsigned* __restrict__ bits) {
  int idx = blockIdx.x * blockDim.x + threadIdx.x;      // one thread per 32 cols
  const int WPR = 64;                                   // words per row (2048 bits)
  if (idx >= Bn * TQ * WPR) return;
  int word = idx % WPR;
  long row = idx / WPR;
  unsigned m = 0;
  int cbase = word * 32;
  const float* p = mask + row * (long)TQ;
#pragma unroll 4
  for (int j = 0; j < 32; ++j) {
    int c = cbase + j;
    if (c < TQ && __builtin_nontemporal_load(p + c) != 0.0f) m |= (1u << j);
  }
  bits[idx] = m;
}

// ---------------- proj GEMM: Y = leaky(X @ W + b), X f32 [M x 512], Wt col-major f16 ----------
__global__ void proj_gemm(const float* __restrict__ X, const _Float16* __restrict__ Wt,
                          const float* __restrict__ bias, _Float16* __restrict__ Y, int M) {
  int lane = threadIdx.x & 31, l16 = lane & 15, hi = (lane >> 4) & 1;
  int wid = (blockIdx.x * blockDim.x + threadIdx.x) >> 5;
  const int nStrips = DM / 64;
  int mTiles = (M + 15) >> 4;
  if (wid >= mTiles * nStrips) return;
  int m0 = (wid / nStrips) * 16;
  int n0 = (wid % nStrips) * 64;
  int rA = m0 + l16; if (rA > M - 1) rA = M - 1;
  const float* rowp = X + (long)rA * DM;
  v8f acc[4] = {};
  for (int k0 = 0; k0 < DM; k0 += 32) {
    v16h a = frag_a_row_f32(rowp + k0, hi, DM - k0);
#pragma unroll
    for (int t = 0; t < 4; ++t) {
      v16h b = frag_b_col_f16(Wt + (long)(n0 + t * 16 + l16) * DM, k0, hi);
      acc[t] = wmma_f32(a, b, acc[t]);
    }
  }
#pragma unroll
  for (int r = 0; r < 8; ++r) {
    int row = m0 + hi * 8 + r;
    if (row >= M) continue;
#pragma unroll
    for (int t = 0; t < 4; ++t) {
      int col = n0 + t * 16 + l16;
      float y = acc[t][r] + bias[col];
      y = (y >= 0.0f) ? y : SLOPE * y;
      Y[(long)row * DM + col] = (_Float16)y;
    }
  }
}

// ---------------- head dense: Out = Y[:, h*64 : h*64+64] @ Ht + bias ----------------
// transposeOut==0: Out[((b*H+h)*T + t)*64 + d]   transposeOut==1: Out[((b*H+h)*64 + d)*T + t]
__global__ void head_dense(const _Float16* __restrict__ Y, const _Float16* __restrict__ Ht,
                           const float* __restrict__ bias, _Float16* __restrict__ Out,
                           int Trows, int transposeOut) {
  int lane = threadIdx.x & 31, l16 = lane & 15, hi = (lane >> 4) & 1;
  int h = blockIdx.y;
  int wid = (blockIdx.x * blockDim.x + threadIdx.x) >> 5;
  int M = Bn * Trows;
  int mTiles = (M + 15) >> 4;
  if (wid >= mTiles) return;
  int m0 = wid * 16;
  int rA = m0 + l16; if (rA > M - 1) rA = M - 1;
  const _Float16* rowp = Y + (long)rA * DM + h * DEPTH;
  v8f acc[4] = {};
  for (int k0 = 0; k0 < DEPTH; k0 += 32) {
    v16h a = frag_a_row_f16(rowp + k0, hi);
#pragma unroll
    for (int t = 0; t < 4; ++t) {
      v16h b = frag_b_col_f16(Ht + (long)(t * 16 + l16) * DEPTH, k0, hi);
      acc[t] = wmma_f32(a, b, acc[t]);
    }
  }
#pragma unroll
  for (int r = 0; r < 8; ++r) {
    int row = m0 + hi * 8 + r;
    if (row >= M) continue;
    int b = row / Trows;
    int tt = row % Trows;
#pragma unroll
    for (int t = 0; t < 4; ++t) {
      int d = t * 16 + l16;
      float y = acc[t][r] + bias[d];
      if (transposeOut)
        Out[((long)(b * H + h) * DEPTH + d) * T + tt] = (_Float16)y;
      else
        Out[((long)(b * H + h) * T + tt) * DEPTH + d] = (_Float16)y;
    }
  }
}

// ================= fused scores + softmax + attn@V =================
// One 256-thread block per (16-row tile, bh). Score strip 16x2048 f32 lives in LDS
// (CDNA5: 320KB/workgroup). attn is written exactly once (nontemporal, streaming);
// attn@V consumes the LDS copy, so the 536MB matrix crosses HBM once instead of 4x.
// Mask bits for the block's 16 rows are staged into LDS by the Tensor Data Mover
// (tensor_load_to_lds + s_wait_tensorcnt), completed before the QK^T WMMAs; TDM OOB
// reads return zero, which handles the ragged last row-tile for free.
__global__ void fused_attn(const _Float16* __restrict__ Qh, const _Float16* __restrict__ Kh,
                           const unsigned* __restrict__ mbits, const _Float16* __restrict__ VT,
                           float* __restrict__ attnOut, _Float16* __restrict__ AO) {
  __shared__ float    s_sc[16][T];     // 128 KB score strip (col 2047 = zero pad)
  __shared__ float    s_part[16][64];  // 4 KB  partial acc for K-split reduction
  __shared__ float    s_red[256];
  __shared__ float    s_stat[16];
  __shared__ unsigned s_mb[16][64];    // 4 KB  mask bits for this row tile

  int tid  = threadIdx.x;
  int lane = tid & 31, l16 = lane & 15, hi = (lane >> 4) & 1;
  int w    = tid >> 5;                // wave 0..7
  int m0   = blockIdx.x * 16;
  int bh   = blockIdx.y;
  int b    = bh / H;
  int h    = bh % H;
  const _Float16* Qb = Qh + (long)bh * T * DEPTH;
  const _Float16* Kb = Kh + (long)bh * T * DEPTH;
  const _Float16* Vb = VT + (long)bh * DEPTH * T;   // padded t-stride 2048

  // ---- TDM: stage 16x64-word mask-bit tile into LDS (issued once, by wave 0) ----
#if __has_builtin(__builtin_amdgcn_tensor_load_to_lds) && __has_builtin(__builtin_amdgcn_s_wait_tensorcnt)
  if (w == 0) {
    const unsigned* gsrc = mbits + ((long)b * TQ + m0) * 64;
    unsigned long long ga = (unsigned long long)(uintptr_t)gsrc;   // 57-bit global addr
    unsigned ldsOff = (unsigned)(uintptr_t)&s_mb[0][0];            // LDS aperture: addr[31:0] = offset
    unsigned tdim1  = (unsigned)(TQ - m0);                         // rows past this -> zero-fill
    u32x4 g0;
    g0[0] = 1u;                                                    // count=1, no gather
    g0[1] = ldsOff;                                                // lds_addr
    g0[2] = (unsigned)ga;                                          // global_addr[31:0]
    g0[3] = (unsigned)((ga >> 32) & 0x01FFFFFFu) | (2u << 30);     // global_addr[56:32] | type=2
    i32x8 g1;
    g1[0] = 0x20000;                                   // data_size=4B (bits 17:16 = 2)
    g1[1] = (int)(64u << 16);                          // tensor_dim0[15:0]=64
    g1[2] = (int)((tdim1 & 0xFFFFu) << 16);            // tensor_dim0[31:16]=0 | tensor_dim1[15:0]
    g1[3] = (int)(((tdim1 >> 16) & 0xFFFFu) | (64u << 16)); // tensor_dim1[31:16] | tile_dim0=64
    g1[4] = 16;                                        // tile_dim1=16, tile_dim2=0
    g1[5] = 64;                                        // tensor_dim0_stride[31:0]=64
    g1[6] = 0;
    g1[7] = 0;
    i32x4 z = {0, 0, 0, 0};
#if defined(__clang_major__) && (__clang_major__ >= 23)
    i32x8 z8 = {0, 0, 0, 0, 0, 0, 0, 0};
    __builtin_amdgcn_tensor_load_to_lds(g0, g1, z, z, z8, 0);
#else
    __builtin_amdgcn_tensor_load_to_lds(g0, g1, z, z, 0);
#endif
    __builtin_amdgcn_s_wait_tensorcnt(0);
  }
#else
  // fallback: cooperative copy with zero-fill
  for (int i = tid; i < 16 * 64; i += 256) {
    int rr = i >> 6;
    s_mb[rr][i & 63] = (m0 + rr < TQ) ? mbits[((long)b * TQ + m0 + rr) * 64 + (i & 63)] : 0u;
  }
#endif

  // ---- phase 1: S = relu((Q[1:] @ K[:-1]^T)/8) + mask*NEG_INF -> LDS ----
  int rA = m0 + l16; if (rA > TQ - 1) rA = TQ - 1;
  const _Float16* qrow = Qb + (long)(rA + 1) * DEPTH;   // q rows start at t=1
  v16h a0 = frag_a_row_f16(qrow, hi);
  v16h a1 = frag_a_row_f16(qrow + 32, hi);
  __syncthreads();                                      // mask tile visible to all waves
  for (int s = w; s < 32; s += 8) {                     // 32 n-strips of 64
    int n0 = s * 64;
    v8f acc[4] = {};
#pragma unroll
    for (int t2 = 0; t2 < 4; ++t2) {
      int n = n0 + t2 * 16 + l16; if (n > TQ - 1) n = TQ - 1;  // key t = n
      const _Float16* kcol = Kb + (long)n * DEPTH;
      acc[t2] = wmma_f32(a0, frag_b_col_f16(kcol, 0, hi), acc[t2]);
      acc[t2] = wmma_f32(a1, frag_b_col_f16(kcol, 32, hi), acc[t2]);
    }
#pragma unroll
    for (int r = 0; r < 8; ++r) {
      int lrow = hi * 8 + r;
#pragma unroll
      for (int t2 = 0; t2 < 4; ++t2) {
        int col = n0 + t2 * 16 + l16;
        float sv = acc[t2][r] * 0.125f;                 // 1/sqrt(64)
        sv = (sv > 0.0f) ? sv : 0.0f;                   // relu
        if ((s_mb[lrow][col >> 5] >> (col & 31)) & 1u) sv += NEGINF;
        s_sc[lrow][col] = sv;
      }
    }
  }
  __syncthreads();

  // ---- phase 2: softmax in LDS; write attn once (streaming / non-temporal) ----
  int row = tid >> 4, sub = tid & 15;                   // 16 threads per row
  float m = -3.0e38f;
  for (int c = sub; c < TQ; c += 16) m = fmaxf(m, s_sc[row][c]);
  s_red[tid] = m;
  __syncthreads();
  if (sub == 0) {
    float mm = s_red[row * 16];
#pragma unroll
    for (int j = 1; j < 16; ++j) mm = fmaxf(mm, s_red[row * 16 + j]);
    s_stat[row] = mm;
  }
  __syncthreads();
  float rm = s_stat[row];
  float sum = 0.0f;
  for (int c = sub; c < TQ; c += 16) {
    float e = __expf(s_sc[row][c] - rm);
    s_sc[row][c] = e;
    sum += e;
  }
  s_red[tid] = sum;
  __syncthreads();
  if (sub == 0) {
    float ss = 0.0f;
#pragma unroll
    for (int j = 0; j < 16; ++j) ss += s_red[row * 16 + j];
    s_stat[row] = 1.0f / ss;
  }
  __syncthreads();
  float inv = s_stat[row];
  int grow = m0 + row;
  float* arow = attnOut + ((long)bh * TQ + (grow < TQ ? grow : TQ - 1)) * TQ;
  for (int c = sub; c < TQ; c += 16) {
    float a = s_sc[row][c] * inv;
    s_sc[row][c] = a;
    if (grow < TQ) __builtin_nontemporal_store(a, arow + c);
  }
  if (sub == 0) s_sc[row][TQ] = 0.0f;                   // zero the pad column
  __syncthreads();

  // ---- phase 3: AO = attn @ V from LDS; K split across wave-halves ----
  int ntile = w & 3, kh = w >> 2;
  int n0 = ntile * 16;                                  // d-subtile
  const _Float16* vcol = Vb + (long)(n0 + l16) * T;
  v8f acc = {};
  for (int k0 = kh * 1024; k0 < kh * 1024 + 1024; k0 += 32) {
    v16h a;
#pragma unroll
    for (int j = 0; j < 8; ++j) {                       // A from LDS (ds_load), f32->f16
      a[j]     = (_Float16)s_sc[l16][k0 + hi * 8 + j];
      a[j + 8] = (_Float16)s_sc[l16][k0 + 16 + hi * 8 + j];
    }
    acc = wmma_f32(a, frag_b_col_f16(vcol, k0, hi), acc);
  }
  if (kh == 1) {
#pragma unroll
    for (int r = 0; r < 8; ++r) s_part[hi * 8 + r][n0 + l16] = acc[r];
  }
  __syncthreads();
  if (kh == 0) {
#pragma unroll
    for (int r = 0; r < 8; ++r) {
      int lrow = hi * 8 + r;
      int growr = m0 + lrow;
      if (growr < TQ) {
        float yv = acc[r] + s_part[lrow][n0 + l16];
        AO[((long)(b * TQ + growr)) * DM + h * DEPTH + n0 + l16] = (_Float16)yv;
      }
    }
  }
}

// ---------------- final GEMM: Out = AO @ wo + bo (f32 output) ----------------
__global__ void out_gemm(const _Float16* __restrict__ A, const _Float16* __restrict__ Wt,
                         const float* __restrict__ bias, float* __restrict__ Out, int M) {
  int lane = threadIdx.x & 31, l16 = lane & 15, hi = (lane >> 4) & 1;
  int wid = (blockIdx.x * blockDim.x + threadIdx.x) >> 5;
  const int nStrips = DM / 64;
  int mTiles = (M + 15) >> 4;
  if (wid >= mTiles * nStrips) return;
  int m0 = (wid / nStrips) * 16;
  int n0 = (wid % nStrips) * 64;
  int rA = m0 + l16; if (rA > M - 1) rA = M - 1;
  const _Float16* rowp = A + (long)rA * DM;
  v8f acc[4] = {};
  for (int k0 = 0; k0 < DM; k0 += 32) {
    v16h a = frag_a_row_f16(rowp + k0, hi);
#pragma unroll
    for (int t = 0; t < 4; ++t) {
      v16h b = frag_b_col_f16(Wt + (long)(n0 + t * 16 + l16) * DM, k0, hi);
      acc[t] = wmma_f32(a, b, acc[t]);
    }
  }
#pragma unroll
  for (int r = 0; r < 8; ++r) {
    int row = m0 + hi * 8 + r;
    if (row >= M) continue;
#pragma unroll
    for (int t = 0; t < 4; ++t) {
      int col = n0 + t * 16 + l16;
      Out[(long)row * DM + col] = acc[t][r] + bias[col];
    }
  }
}

// ---------------- launch ----------------
extern "C" void kernel_launch(void* const* d_in, const int* in_sizes, int n_in,
                              void* d_out, int out_size, void* d_ws, size_t ws_size,
                              hipStream_t stream) {
  const float* v_in = (const float*)d_in[0];
  const float* k_in = (const float*)d_in[1];
  const float* q_in = (const float*)d_in[2];
  const float* mask = (const float*)d_in[3];
  const float* wq = (const float*)d_in[4];  const float* bq  = (const float*)d_in[5];
  const float* wk = (const float*)d_in[6];  const float* bk  = (const float*)d_in[7];
  const float* wv = (const float*)d_in[8];  const float* bv  = (const float*)d_in[9];
  const float* hq = (const float*)d_in[10]; const float* bhq = (const float*)d_in[11];
  const float* hk = (const float*)d_in[12]; const float* bhk = (const float*)d_in[13];
  const float* hv = (const float*)d_in[14]; const float* bhv = (const float*)d_in[15];
  const float* wo = (const float*)d_in[16]; const float* bo  = (const float*)d_in[17];

  float* out  = (float*)d_out;                         // [B, TQ, DM]
  float* attn = out + (long)Bn * TQ * DM;              // [B, H, TQ, TQ]

  // workspace carve-up
  _Float16* p = (_Float16*)d_ws;
  _Float16* wqT = p; p += (long)DM * DM;
  _Float16* wkT = p; p += (long)DM * DM;
  _Float16* wvT = p; p += (long)DM * DM;
  _Float16* woT = p; p += (long)DM * DM;
  _Float16* hqT = p; p += DEPTH * DEPTH;
  _Float16* hkT = p; p += DEPTH * DEPTH;
  _Float16* hvT = p; p += DEPTH * DEPTH;
  _Float16* Yq  = p; p += (long)Bn * T * DM;
  _Float16* Yk  = p; p += (long)Bn * T * DM;
  _Float16* Yv  = p; p += (long)Bn * TQ * DM;
  _Float16* Qh  = p; p += (long)Bn * H * T * DEPTH;
  _Float16* Kh  = p; p += (long)Bn * H * T * DEPTH;
  _Float16* VTr = p; p += (long)Bn * H * DEPTH * T;    // padded t-stride 2048
  _Float16* AO  = p; p += (long)Bn * TQ * DM;
  unsigned* mbits = (unsigned*)p;                      // Bn*TQ*64 words (~2MB)

  // 1) weights -> f16 col-major; mask -> bitmask
  w_to_f16t<<<(DM * DM + 255) / 256, 256, 0, stream>>>(wq, wqT, DM, DM);
  w_to_f16t<<<(DM * DM + 255) / 256, 256, 0, stream>>>(wk, wkT, DM, DM);
  w_to_f16t<<<(DM * DM + 255) / 256, 256, 0, stream>>>(wv, wvT, DM, DM);
  w_to_f16t<<<(DM * DM + 255) / 256, 256, 0, stream>>>(wo, woT, DM, DM);
  w_to_f16t<<<(DEPTH * DEPTH + 255) / 256, 256, 0, stream>>>(hq, hqT, DEPTH, DEPTH);
  w_to_f16t<<<(DEPTH * DEPTH + 255) / 256, 256, 0, stream>>>(hk, hkT, DEPTH, DEPTH);
  w_to_f16t<<<(DEPTH * DEPTH + 255) / 256, 256, 0, stream>>>(hv, hvT, DEPTH, DEPTH);
  mask_to_bits<<<(Bn * TQ * 64 + 255) / 256, 256, 0, stream>>>(mask, mbits);

  // 2) big projections with fused bias + leaky-relu
  int Mqk = Bn * T;    // 8192
  int Mv  = Bn * TQ;   // 8188
  auto blocks4 = [](int strips) { return (strips + 3) / 4; };  // 4 waves / 128-thread block
  int stripsQK = ((Mqk + 15) / 16) * (DM / 64);
  int stripsV  = ((Mv + 15) / 16) * (DM / 64);
  proj_gemm<<<blocks4(stripsQK), 128, 0, stream>>>(q_in, wqT, bq, Yq, Mqk);
  proj_gemm<<<blocks4(stripsQK), 128, 0, stream>>>(k_in, wkT, bk, Yk, Mqk);
  proj_gemm<<<blocks4(stripsV),  128, 0, stream>>>(v_in, wvT, bv, Yv, Mv);

  // 3) shared per-head 64x64 dense (+bias); V written transposed
  {
    dim3 g((((Mqk + 15) / 16) + 3) / 4, H);
    head_dense<<<g, 128, 0, stream>>>(Yq, hqT, bhq, Qh, T, 0);
    head_dense<<<g, 128, 0, stream>>>(Yk, hkT, bhk, Kh, T, 0);
    dim3 gv((((Mv + 15) / 16) + 3) / 4, H);
    head_dense<<<gv, 128, 0, stream>>>(Yv, hvT, bhv, VTr, TQ, 1);
  }

  // 4) fused scores -> softmax -> attn write (1x) -> attn@V  (TDM-staged mask bits)
  {
    dim3 g((TQ + 15) / 16, Bn * H);   // (128, 32)
    fused_attn<<<g, 256, 0, stream>>>(Qh, Kh, mbits, VTr, attn, AO);
  }

  // 5) final projection + bias -> f32 output
  out_gemm<<<blocks4(stripsV), 128, 0, stream>>>(AO, woT, bo, out, Mv);
}
